// Grok1DecoderLayer_76381698392614
// MI455X (gfx1250) — matrix operand order repaired and checked by reading, not compiled, not run on hardware
//
#include <hip/hip_runtime.h>
#include <hip/hip_bf16.h>
#include <math.h>

// ---------------- problem constants ----------------
#define TB   2
#define SSQ  1024
#define HHD  1024
#define NHH  16
#define HDD  64
#define IID  2048
#define EEX  8
#define TT   (TB * SSQ)      // 2048 tokens
#define DDm  (NHH * HDD)     // 1024
#define TWO_I (2 * IID)      // 4096
#define NSLOT (TT * 2)       // 4096 (every token picks exactly 2 experts)
#define NEGV  (-1e30f)
#define EPSV  1e-8f
#define ALPHA_ 1.702f
#define LIMIT_ 7.0f

typedef __attribute__((ext_vector_type(16))) _Float16 v16h;
typedef __attribute__((ext_vector_type(8)))  float    v8f;
typedef __attribute__((ext_vector_type(4)))  int      v4i_;

#define WMMA_F16(a,b,c) __builtin_amdgcn_wmma_f32_16x16x32_f16(false,(a),false,(b),(short)0,(c),false,false)

// ---- CDNA5 async global->LDS copy (guarded; falls back to sync staging) ----
#if __has_builtin(__builtin_amdgcn_global_load_async_to_lds_b128)
#define ASYNC_LDS 1
__device__ __forceinline__ void async_copy16(const void* g, void* l) {
  __builtin_amdgcn_global_load_async_to_lds_b128((v4i_*)g, (v4i_*)l, 0, 0);
}
__device__ __forceinline__ void async_wait0() {
  asm volatile("s_wait_asynccnt 0" ::: "memory");
}
#else
#define ASYNC_LDS 0
#endif

__device__ __forceinline__ int lane32() { return threadIdx.x & 31; }

// ---- WMMA fragment loaders (CDNA5 16x16x32 f16 layouts) ----
// A 16(M)x32(K): lane<16 -> M=lane, K {k0..k0+7, k0+16..k0+23}; lane>=16 -> M=lane-16, K {+8.., +24..}
__device__ __forceinline__ v16h frag_a_h(const _Float16* src, int ld, int k0) {
  int l = lane32();
  const _Float16* p = src + (size_t)(l & 15) * ld + k0 + ((l >> 4) << 3);
  v16h a;
  #pragma unroll
  for (int i = 0; i < 8; ++i) { a[i] = p[i]; a[8 + i] = p[16 + i]; }
  return a;
}
__device__ __forceinline__ v16h frag_a_f32(const float* src, int ld, int k0) {
  int l = lane32();
  const float* p = src + (size_t)(l & 15) * ld + k0 + ((l >> 4) << 3);
  v16h a;
  #pragma unroll
  for (int i = 0; i < 8; ++i) { a[i] = (_Float16)p[i]; a[8 + i] = (_Float16)p[16 + i]; }
  return a;
}
// B 32(K)x16(N) read from row-major [N][K] storage (A@W^T pattern): lane<16 -> N=lane, K {k0..k0+7, k0+16..k0+23}; lane>=16 -> {+8.., +24..}
__device__ __forceinline__ v16h frag_bt_h(const _Float16* src, int ld, int k0) {
  int l = lane32();
  const _Float16* p = src + (size_t)(l & 15) * ld + k0 + ((l >> 4) << 3);
  v16h b;
  #pragma unroll
  for (int i = 0; i < 8; ++i) { b[i] = p[i]; b[8 + i] = p[16 + i]; }
  return b;
}
__device__ __forceinline__ v16h frag_bt_f32(const float* src, int ld, int k0) {
  int l = lane32();
  const float* p = src + (size_t)(l & 15) * ld + k0 + ((l >> 4) << 3);
  v16h b;
  #pragma unroll
  for (int i = 0; i < 8; ++i) { b[i] = (_Float16)p[i]; b[8 + i] = (_Float16)p[16 + i]; }
  return b;
}

// ---- block reduction helper (wave32-aware) ----
__device__ __forceinline__ float block_reduce_sum256(float v, float* red) {
  #pragma unroll
  for (int off = 16; off; off >>= 1) v += __shfl_xor(v, off, 32);
  if ((threadIdx.x & 31) == 0) red[threadIdx.x >> 5] = v;
  __syncthreads();
  if (threadIdx.x < 32) {
    float r = (threadIdx.x < 8) ? red[threadIdx.x] : 0.f;
    #pragma unroll
    for (int off = 4; off; off >>= 1) r += __shfl_xor(r, off, 8);
    if (threadIdx.x == 0) red[0] = r;
  }
  __syncthreads();
  return red[0];
}

// ================= kernels =================

__global__ __launch_bounds__(256) void zero_kernel(float* moe, int* counts, int* fills, int n) {
  int i = blockIdx.x * 256 + threadIdx.x;
  if (i < n) moe[i] = 0.f;
  if (i < EEX) { counts[i] = 0; fills[i] = 0; }
}

// out_f16[row] = w * x * sqrt(mean(x^2)+eps)   (faithful to reference formula)
__global__ __launch_bounds__(256) void rmsnorm_f16_kernel(const float* __restrict__ x,
                                                          const float* __restrict__ w,
                                                          _Float16* __restrict__ out) {
  __shared__ float red[8];
  int row = blockIdx.x;
  const float* xr = x + (size_t)row * HHD;
  float ss = 0.f;
  for (int i = threadIdx.x; i < HHD; i += 256) { float v = xr[i]; ss += v * v; }
  float tot = block_reduce_sum256(ss, red);
  float scale = sqrtf(tot / (float)HHD + EPSV);
  for (int i = threadIdx.x; i < HHD; i += 256)
    out[(size_t)row * HHD + i] = (_Float16)(w[i] * xr[i] * scale);
}

// hout = hin + w * y * sqrt(mean(y^2)+eps)
__global__ __launch_bounds__(256) void resid_rmsnorm_kernel(const float* __restrict__ hin,
                                                            const float* __restrict__ y,
                                                            const float* __restrict__ w,
                                                            float* __restrict__ hout) {
  __shared__ float red[8];
  int row = blockIdx.x;
  const float* yr = y + (size_t)row * HHD;
  float ss = 0.f;
  for (int i = threadIdx.x; i < HHD; i += 256) { float v = yr[i]; ss += v * v; }
  float tot = block_reduce_sum256(ss, red);
  float scale = sqrtf(tot / (float)HHD + EPSV);
  for (int i = threadIdx.x; i < HHD; i += 256) {
    size_t idx = (size_t)row * HHD + i;
    hout[idx] = hin[idx] + w[i] * yr[i] * scale;
  }
}

// C[M,N] = A_f16[M,K] @ W_f32[N,K]^T  -- 4 waves/block, each wave one 16x16 tile column
__global__ __launch_bounds__(128) void gemm_nt_kernel(const _Float16* __restrict__ A,
                                                      const float* __restrict__ W,
                                                      float* __restrict__ C,
                                                      int N, int K) {
  int wid = threadIdx.x >> 5;
  int n0 = blockIdx.x << 4;
  int m0 = (blockIdx.y << 6) + (wid << 4);
  const _Float16* Abase = A + (size_t)m0 * K;
  const float* Wbase = W + (size_t)n0 * K;
  v8f acc = {};
  for (int k0 = 0; k0 < K; k0 += 32) {
    __builtin_prefetch(Wbase + (size_t)(lane32() & 15) * K + k0 + 64, 0, 1);
    v16h a = frag_a_h(Abase, K, k0);
    v16h b = frag_bt_f32(Wbase, K, k0);
    acc = WMMA_F16(a, b, acc);
  }
  int l = lane32();
  int cm = m0 + ((l >> 4) << 3);
  int cn = n0 + (l & 15);
  #pragma unroll
  for (int i = 0; i < 8; ++i) C[(size_t)(cm + i) * N + cn] = acc[i];
}

// RoPE + repack: q,k -> (b,h,s,d) f16;  v -> (b,h,d,s) f16 (transposed for P@V fragments)
__global__ __launch_bounds__(256) void rope_pack_kernel(const float* __restrict__ qf,
                                                        const float* __restrict__ kf,
                                                        const float* __restrict__ vf,
                                                        const float* __restrict__ cosb,
                                                        const float* __restrict__ sinb,
                                                        _Float16* __restrict__ qh,
                                                        _Float16* __restrict__ kh,
                                                        _Float16* __restrict__ vt) {
  int idx = blockIdx.x * 256 + threadIdx.x;   // over TT * NHH * 32
  if (idx >= TT * NHH * 32) return;
  int d2 = idx & 31;
  int h  = (idx >> 5) & (NHH - 1);
  int t  = idx >> 9;
  int b = t >> 10, s = t & (SSQ - 1);
  float c  = cosb[(size_t)t * 32 + d2];
  float sn = sinb[(size_t)t * 32 + d2];
  size_t src = (size_t)t * DDm + h * HDD + d2;
  size_t dst = (((size_t)b * NHH + h) * SSQ + s) * HDD + d2;
  float q1 = qf[src], q2 = qf[src + 32];
  qh[dst]      = (_Float16)(q1 * c - q2 * sn);
  qh[dst + 32] = (_Float16)(q2 * c + q1 * sn);
  float k1 = kf[src], k2 = kf[src + 32];
  kh[dst]      = (_Float16)(k1 * c - k2 * sn);
  kh[dst + 32] = (_Float16)(k2 * c + k1 * sn);
  size_t dv = (((size_t)b * NHH + h) * HDD + d2) * SSQ + s;
  vt[dv]              = (_Float16)vf[src];
  vt[dv + 32 * SSQ]   = (_Float16)vf[src + 32];
}

// Fused attention: one block per (32 query rows, head, batch). Scores in dynamic LDS.
__global__ __launch_bounds__(256) void attn_kernel(const _Float16* __restrict__ qh,
                                                   const _Float16* __restrict__ kh,
                                                   const _Float16* __restrict__ vt,
                                                   const float* __restrict__ sinks,
                                                   _Float16* __restrict__ attn) {
  extern __shared__ float sc[];              // [32][1032]
  const int LD = 1032;
  int qb = blockIdx.x, h = blockIdx.y, b = blockIdx.z;
  int q0 = qb * 32;
  const _Float16* qbase = qh + (((size_t)b * NHH + h) * SSQ) * HDD;
  const _Float16* kbase = kh + (((size_t)b * NHH + h) * SSQ) * HDD;
  const _Float16* vbase = vt + (((size_t)b * NHH + h) * HDD) * SSQ;
  int wid = threadIdx.x >> 5, l = lane32();
  // ---- pass 1: scores via WMMA, scaled + causal-masked into LDS ----
  // q fragments are invariant across key tiles: load once (2 m-subtiles x 2 K-steps)
  v16h aq[2][2];
  #pragma unroll
  for (int mt = 0; mt < 2; ++mt)
    #pragma unroll
    for (int kk = 0; kk < 2; ++kk)
      aq[mt][kk] = frag_a_h(qbase + (size_t)(q0 + mt * 16) * HDD, HDD, kk * 32);
  int ntmax = (q0 + 31) >> 4;                // only keys t <= q0+31 needed (SW >= S -> pure causal)
  for (int nt = wid; nt <= ntmax; nt += 8) {
    v16h bk0 = frag_bt_h(kbase + (size_t)nt * 16 * HDD, HDD, 0);
    v16h bk1 = frag_bt_h(kbase + (size_t)nt * 16 * HDD, HDD, 32);
    #pragma unroll
    for (int mt = 0; mt < 2; ++mt) {
      v8f acc = {};
      acc = WMMA_F16(aq[mt][0], bk0, acc);
      acc = WMMA_F16(aq[mt][1], bk1, acc);
      int sl = mt * 16 + ((l >> 4) << 3);
      int t  = nt * 16 + (l & 15);
      #pragma unroll
      for (int i = 0; i < 8; ++i) {
        int sg = q0 + sl + i;
        sc[(size_t)(sl + i) * LD + t] = (t <= sg) ? acc[i] * 0.125f : NEGV;
      }
    }
  }
  __syncthreads();
  // ---- pass 2: softmax with sink (8 threads per row) ----
  int kmax = q0 + 32;
  {
    int r = threadIdx.x >> 3, j = threadIdx.x & 7;
    float* row = sc + (size_t)r * LD;
    float snk = sinks[h];
    float mx = -3.0e38f;
    for (int t = j; t < kmax; t += 8) mx = fmaxf(mx, row[t]);
    #pragma unroll
    for (int off = 4; off; off >>= 1) mx = fmaxf(mx, __shfl_xor(mx, off, 8));
    mx = fmaxf(mx, snk);
    float ls = 0.f;
    for (int t = j; t < kmax; t += 8) { float p = __expf(row[t] - mx); row[t] = p; ls += p; }
    #pragma unroll
    for (int off = 4; off; off >>= 1) ls += __shfl_xor(ls, off, 8);
    ls += __expf(snk - mx);
    float inv = 1.f / ls;
    for (int t = j; t < kmax; t += 8) row[t] *= inv;
  }
  __syncthreads();
  // ---- pass 3: P @ V via WMMA (A from LDS f32, B from transposed V f16) ----
  {
    int mt = wid >> 2;
    int nd = (wid & 3) << 4;
    v8f acc = {};
    for (int k0 = 0; k0 < kmax; k0 += 32) {
      v16h a = frag_a_f32(sc + (size_t)mt * 16 * LD, LD, k0);
      v16h bb = frag_bt_h(vbase + (size_t)nd * SSQ, SSQ, k0);
      acc = WMMA_F16(a, bb, acc);
    }
    int s0 = q0 + mt * 16 + ((l >> 4) << 3);
    int cn = h * HDD + nd + (l & 15);
    size_t trow = (size_t)b * SSQ;
    #pragma unroll
    for (int i = 0; i < 8; ++i)
      attn[(trow + s0 + i) * DDm + cn] = (_Float16)acc[i];
  }
}

// Router: block per token, top-2 + softmax, count assignments
__global__ __launch_bounds__(256) void router_kernel(const _Float16* __restrict__ xn,
                                                     const float* __restrict__ rw,
                                                     const float* __restrict__ rb,
                                                     int* __restrict__ counts,
                                                     int* __restrict__ ti, float* __restrict__ tw) {
  int t = blockIdx.x;
  const _Float16* xr = xn + (size_t)t * HHD;
  float acc[EEX];
  #pragma unroll
  for (int e = 0; e < EEX; ++e) acc[e] = 0.f;
  for (int i = threadIdx.x; i < HHD; i += 256) {
    float xv = (float)xr[i];
    #pragma unroll
    for (int e = 0; e < EEX; ++e) acc[e] += xv * rw[(size_t)e * HHD + i];
  }
  __shared__ float lg[EEX];
  if (threadIdx.x < EEX) lg[threadIdx.x] = 0.f;
  __syncthreads();
  #pragma unroll
  for (int e = 0; e < EEX; ++e) {
    float v = acc[e];
    #pragma unroll
    for (int off = 16; off; off >>= 1) v += __shfl_xor(v, off, 32);
    if ((threadIdx.x & 31) == 0) atomicAdd(&lg[e], v);
  }
  __syncthreads();
  if (threadIdx.x == 0) {
    float lv[EEX];
    #pragma unroll
    for (int e = 0; e < EEX; ++e) lv[e] = lg[e] + rb[e];
    int b0 = 0; float v0 = lv[0];
    for (int e = 1; e < EEX; ++e) if (lv[e] > v0) { v0 = lv[e]; b0 = e; }
    int b1 = (b0 == 0) ? 1 : 0; float v1 = lv[b1];
    for (int e = 0; e < EEX; ++e) if (e != b0 && lv[e] > v1) { v1 = lv[e]; b1 = e; }
    float e1 = __expf(v1 - v0);
    float inv = 1.f / (1.f + e1);
    ti[2 * t] = b0; ti[2 * t + 1] = b1;
    tw[2 * t] = inv; tw[2 * t + 1] = e1 * inv;
    atomicAdd(&counts[b0], 1);
    atomicAdd(&counts[b1], 1);
  }
}

__global__ void scan_kernel(const int* __restrict__ counts, int* __restrict__ base) {
  if (threadIdx.x == 0) {
    int s = 0;
    for (int e = 0; e < EEX; ++e) { base[e] = s; s += counts[e]; }
    base[EEX] = s;
  }
}

__global__ __launch_bounds__(256) void fill_kernel(const int* __restrict__ ti,
                                                   const float* __restrict__ tw,
                                                   const int* __restrict__ base,
                                                   int* __restrict__ fill,
                                                   int* __restrict__ tok, float* __restrict__ twg) {
  int t = blockIdx.x * 256 + threadIdx.x;
  if (t >= TT) return;
  #pragma unroll
  for (int k = 0; k < 2; ++k) {
    int e = ti[2 * t + k];
    int pos = base[e] + atomicAdd(&fill[e], 1);
    tok[pos] = t;
    twg[pos] = tw[2 * t + k];
  }
}

// ---- shared B-tile staging for the expert GEMMs ----
// Stages a 32(K) x 64(N) tile of a row-major [K][N] f32 weight matrix.
#if ASYNC_LDS
// async path: raw f32 tile [k][n], row padded to 65 floats so per-lane column
// reads (16 k-values, fixed n) hit 16 distinct banks.
#define BTILE_DECL __shared__ float Bs[32][65]
__device__ __forceinline__ void stage_btile(const float* W, int ldw, int k0, int n0b,
                                            float (*Bs)[65]) {
  int kk = threadIdx.x >> 3;
  int nn = (threadIdx.x & 7) << 3;
  const float* p = W + (size_t)(k0 + kk) * ldw + n0b + nn;
  async_copy16(p,     &Bs[kk][nn]);
  async_copy16(p + 4, &Bs[kk][nn + 4]);
  async_wait0();
}
__device__ __forceinline__ v16h frag_btile(const float (*Bs)[65], int nsub) {
  int l = lane32();
  int n = nsub + (l & 15);
  int kb = (l >> 4) << 3;
  v16h b;
  #pragma unroll
  for (int i = 0; i < 8; ++i) { b[i] = (_Float16)Bs[kb + i][n]; b[8 + i] = (_Float16)Bs[kb + 16 + i][n]; }
  return b;
}
#else
// sync fallback: convert while staging, transposed [n][k] f16 so fragment reads
// are two contiguous 16B runs per lane.
#define BTILE_DECL __shared__ _Float16 Bs[64][40]
__device__ __forceinline__ void stage_btile(const float* W, int ldw, int k0, int n0b,
                                            _Float16 (*Bs)[40]) {
  int kk = threadIdx.x >> 3;
  int nn = (threadIdx.x & 7) << 3;
  const float* p = W + (size_t)(k0 + kk) * ldw + n0b + nn;
  #pragma unroll
  for (int i = 0; i < 8; ++i) Bs[nn + i][kk] = (_Float16)p[i];
}
__device__ __forceinline__ v16h frag_btile(const _Float16 (*Bs)[40], int nsub) {
  int l = lane32();
  const _Float16* pb = &Bs[nsub + (l & 15)][(l >> 4) << 3];
  v16h b;
  #pragma unroll
  for (int i = 0; i < 8; ++i) { b[i] = pb[i]; b[8 + i] = pb[16 + i]; }
  return b;
}
#endif

// gate_up GEMM: C[m,f] = sum_h X[tok[m],h] * gup[e][h][f] + bias; 64x64 tile
__global__ __launch_bounds__(256) void moe_gemm1_kernel(const _Float16* __restrict__ X,
                                                        const float* __restrict__ gup,
                                                        const float* __restrict__ gub,
                                                        const int* __restrict__ tok,
                                                        const int* __restrict__ base,
                                                        const int* __restrict__ counts,
                                                        float* __restrict__ gu) {
  int e = blockIdx.z;
  int cnt = counts[e];
  int mt0 = blockIdx.y << 6;
  if (mt0 >= cnt) return;
  int b0 = base[e];
  int n0b = blockIdx.x << 6;
  const float* W = gup + (size_t)e * HHD * TWO_I;
  BTILE_DECL;
  int wid = threadIdx.x >> 5, l = lane32();
  int msub = (wid >> 2) << 5;
  int nsub = (wid & 3) << 4;
  int am0 = mt0 + msub + (l & 15);
  int am1 = am0 + 16;
  int r0 = tok[b0 + (am0 < cnt ? am0 : cnt - 1)];
  int r1 = tok[b0 + (am1 < cnt ? am1 : cnt - 1)];
  const _Float16* a0p = X + (size_t)r0 * HHD + ((l >> 4) << 3);
  const _Float16* a1p = X + (size_t)r1 * HHD + ((l >> 4) << 3);
  v8f acc0 = {}, acc1 = {};
  for (int k0 = 0; k0 < HHD; k0 += 32) {
    __syncthreads();
    stage_btile(W, TWO_I, k0, n0b, Bs);
    __syncthreads();
    v16h a0, a1;
    const _Float16* pa0 = a0p + k0;
    const _Float16* pa1 = a1p + k0;
    #pragma unroll
    for (int i = 0; i < 8; ++i) {
      a0[i] = pa0[i]; a0[8 + i] = pa0[16 + i];
      a1[i] = pa1[i]; a1[8 + i] = pa1[16 + i];
    }
    v16h b = frag_btile(Bs, nsub);
    acc0 = WMMA_F16(a0, b, acc0);
    acc1 = WMMA_F16(a1, b, acc1);
  }
  int cn = n0b + nsub + (l & 15);
  float bias = gub[(size_t)e * TWO_I + cn];
  int cm = mt0 + msub + ((l >> 4) << 3);
  #pragma unroll
  for (int i = 0; i < 8; ++i) {
    if (cm + i < cnt)      gu[(size_t)(b0 + cm + i) * TWO_I + cn]      = acc0[i] + bias;
    if (cm + 16 + i < cnt) gu[(size_t)(b0 + cm + 16 + i) * TWO_I + cn] = acc1[i] + bias;
  }
}

// GLU activation over all 4096 slots
__global__ __launch_bounds__(256) void act_kernel(const float* __restrict__ gu,
                                                  _Float16* __restrict__ act) {
  size_t idx = (size_t)blockIdx.x * 256 + threadIdx.x;   // NSLOT * IID
  if (idx >= (size_t)NSLOT * IID) return;
  size_t r = idx >> 11;
  int i = (int)(idx & (IID - 1));
  float gate = gu[r * TWO_I + 2 * i];
  float up   = gu[r * TWO_I + 2 * i + 1];
  gate = fminf(gate, LIMIT_);
  up   = fminf(fmaxf(up, -LIMIT_), LIMIT_);
  float glu = gate / (1.f + __expf(-ALPHA_ * gate));
  act[idx] = (_Float16)((up + 1.f) * glu);
}

// down GEMM + weighted scatter: moe[tok[m]] += twg[m] * (act[m] @ down[e] + bias)
__global__ __launch_bounds__(256) void moe_gemm2_kernel(const _Float16* __restrict__ act,
                                                        const float* __restrict__ dw,
                                                        const float* __restrict__ db,
                                                        const int* __restrict__ tok,
                                                        const float* __restrict__ twg,
                                                        const int* __restrict__ base,
                                                        const int* __restrict__ counts,
                                                        float* __restrict__ moe) {
  int e = blockIdx.z;
  int cnt = counts[e];
  int mt0 = blockIdx.y << 6;
  if (mt0 >= cnt) return;
  int b0 = base[e];
  int n0b = blockIdx.x << 6;
  const float* W = dw + (size_t)e * IID * HHD;
  BTILE_DECL;
  int wid = threadIdx.x >> 5, l = lane32();
  int msub = (wid >> 2) << 5;
  int nsub = (wid & 3) << 4;
  int am0 = mt0 + msub + (l & 15);
  int am1 = am0 + 16;
  int r0 = b0 + (am0 < cnt ? am0 : cnt - 1);
  int r1 = b0 + (am1 < cnt ? am1 : cnt - 1);
  const _Float16* a0p = act + (size_t)r0 * IID + ((l >> 4) << 3);
  const _Float16* a1p = act + (size_t)r1 * IID + ((l >> 4) << 3);
  v8f acc0 = {}, acc1 = {};
  for (int k0 = 0; k0 < IID; k0 += 32) {
    __syncthreads();
    stage_btile(W, HHD, k0, n0b, Bs);
    __syncthreads();
    v16h a0, a1;
    const _Float16* pa0 = a0p + k0;
    const _Float16* pa1 = a1p + k0;
    #pragma unroll
    for (int i = 0; i < 8; ++i) {
      a0[i] = pa0[i]; a0[8 + i] = pa0[16 + i];
      a1[i] = pa1[i]; a1[8 + i] = pa1[16 + i];
    }
    v16h b = frag_btile(Bs, nsub);
    acc0 = WMMA_F16(a0, b, acc0);
    acc1 = WMMA_F16(a1, b, acc1);
  }
  int cn = n0b + nsub + (l & 15);
  float bias = db[(size_t)e * HHD + cn];
  int cm = mt0 + msub + ((l >> 4) << 3);
  #pragma unroll
  for (int i = 0; i < 8; ++i) {
    int m = cm + i;
    if (m < cnt) {
      int slot = b0 + m;
      atomicAdd(&moe[(size_t)tok[slot] * HHD + cn], twg[slot] * (acc0[i] + bias));
    }
    m = cm + 16 + i;
    if (m < cnt) {
      int slot = b0 + m;
      atomicAdd(&moe[(size_t)tok[slot] * HHD + cn], twg[slot] * (acc1[i] + bias));
    }
  }
}

// ================= host side =================
extern "C" void kernel_launch(void* const* d_in, const int* in_sizes, int n_in,
                              void* d_out, int out_size, void* d_ws, size_t ws_size,
                              hipStream_t stream) {
  const float* hidden  = (const float*)d_in[0];
  const float* cosb    = (const float*)d_in[1];
  const float* sinb    = (const float*)d_in[2];
  const float* wq      = (const float*)d_in[3];
  const float* wk      = (const float*)d_in[4];
  const float* wv      = (const float*)d_in[5];
  const float* wo      = (const float*)d_in[6];
  const float* sinks   = (const float*)d_in[7];
  const float* pre_a   = (const float*)d_in[8];
  const float* post_a  = (const float*)d_in[9];
  const float* pre_m   = (const float*)d_in[10];
  const float* post_m  = (const float*)d_in[11];
  const float* rw      = (const float*)d_in[12];
  const float* rb      = (const float*)d_in[13];
  const float* gup     = (const float*)d_in[14];
  const float* gub     = (const float*)d_in[15];
  const float* dwn     = (const float*)d_in[16];
  const float* dwb     = (const float*)d_in[17];
  float* outp = (float*)d_out;

  char* w = (char*)d_ws;
  size_t off = 0;
  auto alloc = [&](size_t bytes) { char* p = w + off; off = (off + bytes + 255) & ~(size_t)255; return p; };
  _Float16* xn1   = (_Float16*)alloc((size_t)TT * HHD * 2);
  float*    qf    = (float*)   alloc((size_t)TT * DDm * 4);
  float*    kf    = (float*)   alloc((size_t)TT * DDm * 4);
  float*    vf    = (float*)   alloc((size_t)TT * DDm * 4);
  _Float16* qh    = (_Float16*)alloc((size_t)TT * DDm * 2);
  _Float16* kh    = (_Float16*)alloc((size_t)TT * DDm * 2);
  _Float16* vt    = (_Float16*)alloc((size_t)TT * DDm * 2);
  _Float16* attn  = (_Float16*)alloc((size_t)TT * DDm * 2);
  float*    aof   = (float*)   alloc((size_t)TT * HHD * 4);
  float*    h2    = (float*)   alloc((size_t)TT * HHD * 4);
  _Float16* xn2   = (_Float16*)alloc((size_t)TT * HHD * 2);
  int*      ti    = (int*)     alloc((size_t)TT * 2 * 4);
  float*    tw    = (float*)   alloc((size_t)TT * 2 * 4);
  int*      counts= (int*)     alloc(256);
  int*      basep = (int*)     alloc(256);
  int*      fills = (int*)     alloc(256);
  int*      tok   = (int*)     alloc((size_t)NSLOT * 4);
  float*    twg   = (float*)   alloc((size_t)NSLOT * 4);
  float*    gu    = (float*)   alloc((size_t)NSLOT * TWO_I * 4);
  _Float16* acth  = (_Float16*)alloc((size_t)NSLOT * IID * 2);
  float*    moe   = (float*)   alloc((size_t)TT * HHD * 4);
  (void)ws_size; (void)n_in; (void)in_sizes; (void)out_size;

  // 0) zero scratch accumulators
  zero_kernel<<<(TT * HHD + 255) / 256, 256, 0, stream>>>(moe, counts, fills, TT * HHD);
  // 1) pre-attn norm
  rmsnorm_f16_kernel<<<TT, 256, 0, stream>>>(hidden, pre_a, xn1);
  // 2) Q/K/V projections (WMMA)
  gemm_nt_kernel<<<dim3(DDm / 16, TT / 64), 128, 0, stream>>>(xn1, wq, qf, DDm, HHD);
  gemm_nt_kernel<<<dim3(DDm / 16, TT / 64), 128, 0, stream>>>(xn1, wk, kf, DDm, HHD);
  gemm_nt_kernel<<<dim3(DDm / 16, TT / 64), 128, 0, stream>>>(xn1, wv, vf, DDm, HHD);
  // 3) RoPE + repack (V transposed per head)
  rope_pack_kernel<<<(TT * NHH * 32 + 255) / 256, 256, 0, stream>>>(qf, kf, vf, cosb, sinb, qh, kh, vt);
  // 4) fused attention, 132KB dynamic LDS per workgroup (CDNA5: 320KB/WGP)
  attn_kernel<<<dim3(SSQ / 32, NHH, TB), 256, 32 * 1032 * 4, stream>>>(qh, kh, vt, sinks, attn);
  // 5) O-projection
  gemm_nt_kernel<<<dim3(HHD / 16, TT / 64), 128, 0, stream>>>(attn, wo, aof, HHD, DDm);
  // 6) residual + post-attn norm; pre-moe norm
  resid_rmsnorm_kernel<<<TT, 256, 0, stream>>>(hidden, aof, post_a, h2);
  rmsnorm_f16_kernel<<<TT, 256, 0, stream>>>(h2, pre_m, xn2);
  // 7) routing
  router_kernel<<<TT, 256, 0, stream>>>(xn2, rw, rb, counts, ti, tw);
  scan_kernel<<<1, 32, 0, stream>>>(counts, basep);
  fill_kernel<<<TT / 256, 256, 0, stream>>>(ti, tw, basep, fills, tok, twg);
  // 8) expert GEMMs (sparse: only selected (token,expert) rows)
  moe_gemm1_kernel<<<dim3(TWO_I / 64, NSLOT / 64, EEX), 256, 0, stream>>>(
      xn2, gup, gub, tok, basep, counts, gu);
  act_kernel<<<(int)(((size_t)NSLOT * IID + 255) / 256), 256, 0, stream>>>(gu, acth);
  moe_gemm2_kernel<<<dim3(HHD / 64, NSLOT / 64, EEX), 256, 0, stream>>>(
      acth, dwn, dwb, tok, twg, basep, counts, moe);
  // 9) final residual + post-moe norm -> output
  resid_rmsnorm_kernel<<<TT, 256, 0, stream>>>(h2, moe, post_m, outp);
}
